// LLTM_88407606821373
// MI455X (gfx1250) — matrix operand order repaired and checked by reading, not compiled, not run on hardware
//
#include <hip/hip_runtime.h>
#include <hip/hip_bf16.h>

// ---------------------------------------------------------------------------
// LLTM fused cell for gfx1250: bf16 WMMA GEMM (X @ W^T + b) + gate epilogue.
// Two 16-row M-tiles per wave; software-pipelined LDS B-fragments.
// ---------------------------------------------------------------------------

typedef __attribute__((ext_vector_type(16))) __bf16 v16bf;
typedef __attribute__((ext_vector_type(2)))  __bf16 v2bf;
typedef __attribute__((ext_vector_type(8)))  float  v8f;
typedef __attribute__((ext_vector_type(4)))  float  v4f;
typedef __attribute__((ext_vector_type(16))) float  v16f;

#define STATE     128
#define IN_FEATS  32
#define KTOT      160            // STATE + IN_FEATS
#define GATES     384            // 3*STATE
#define NT_N      24             // 384 / 16 N-tiles
#define KC        5              // 160 / 32 K-chunks
#define PACK_DWORDS (NT_N * KC * 256)      // 30720 dwords
#define PACK_BYTES  (PACK_DWORDS * 4)      // 122880 bytes
#define WS_DWORDS   (PACK_DWORDS + GATES)  // pack + bias copy = 31104
#define SMEM_BYTES  (PACK_BYTES + GATES * 4)

// Native bf16 pack (lowers to v_cvt_pk_bf16_f32; RNE).
__device__ __forceinline__ unsigned pack2bf(float lo, float hi) {
    v2bf p;
    p[0] = (__bf16)lo;
    p[1] = (__bf16)hi;
    return __builtin_bit_cast(unsigned, p);
}
// Fast activations: v_exp_f32 + v_rcp_f32 (avoid IEEE div expansion).
__device__ __forceinline__ float fast_sigmoid(float x) {
    return __builtin_amdgcn_rcpf(1.0f + __expf(-x));
}
__device__ __forceinline__ float fast_tanh(float x) {
    return 1.0f - 2.0f * __builtin_amdgcn_rcpf(__expf(2.0f * x) + 1.0f);
}

// Swizzle one pack element p -> bf16x2 dword of the WMMA B-operand layout.
// Fragment f = nt*KC + kc holds B tile (K=kc*32..+31, N=nt*16..+15).
// Dense 16-bit B 32x16 layout: lane L -> N = L&15,
//   VGPR j packs K = base+2j, base+2j+1 with base = (j>=4 ? 16 : 0) + (L>=16 ? 8 : 0) - 2*(j&4).
__device__ __forceinline__ unsigned pack_elem(const float* __restrict__ weights,
                                              int p) {
    int f    = p >> 8;
    int r    = p & 255;
    int lane = r >> 3;
    int j    = r & 7;
    int nt   = f / KC;
    int kc   = f - nt * KC;
    int n    = nt * 16 + (lane & 15);
    int k    = kc * 32 + (j >> 2) * 16 + (lane >> 4) * 8 + 2 * (j & 3);
    const float* w = weights + n * KTOT + k;
    return pack2bf(w[0], w[1]);
}

// ---- Kernel A: one-time weight pack + bias copy into workspace ------------
__global__ __launch_bounds__(256) void lltm_pack_weights(
    const float* __restrict__ weights, const float* __restrict__ bias,
    unsigned* __restrict__ ws) {
    int idx = blockIdx.x * 256 + threadIdx.x;
    if (idx < PACK_DWORDS) {
        ws[idx] = pack_elem(weights, idx);
    } else if (idx < WS_DWORDS) {
        ((float*)ws)[idx] = bias[idx - PACK_DWORDS];
    }
}

// ---- Kernel B: fused LLTM cell --------------------------------------------
__global__ __launch_bounds__(256) void lltm_fused_wmma(
    const float* __restrict__ inp,       // [B, 32]
    const float* __restrict__ old_h,     // [B, 128]
    const float* __restrict__ old_cell,  // [B, 128]
    const float* __restrict__ weights,   // [384, 160] row-major
    const float* __restrict__ bias,      // [384]
    const unsigned* __restrict__ packSrc,// pre-packed ws, or nullptr
    float* __restrict__ out,             // [new_h | new_cell], each [B,128]
    int Bn)
{
    extern __shared__ char smem_raw[];
    unsigned* packLds = (unsigned*)smem_raw;                 // 30720 dwords
    float*    biasLds = (float*)(smem_raw + PACK_BYTES);     // 384 floats
    const int tid = threadIdx.x;

    if (packSrc != nullptr) {
        // Fast path: bulk b128 copy ws -> LDS (pack + bias appended).
        const v4f* src = (const v4f*)packSrc;
        v4f*       dst = (v4f*)smem_raw;
        for (int i = tid; i < WS_DWORDS / 4; i += 256) dst[i] = src[i];
    } else {
        // Fallback: compute the pack in-block.
        for (int p = tid; p < PACK_DWORDS; p += 256)
            packLds[p] = pack_elem(weights, p);
        for (int i = tid; i < GATES; i += 256) biasLds[i] = bias[i];
    }
    __syncthreads();

    const int lane   = tid & 31;
    const int wave   = tid >> 5;
    const int hi     = lane >> 4;
    const int l15    = lane & 15;
    const int nPair  = Bn >> 5;                 // 32 rows per pair
    const int gWave  = blockIdx.x * 8 + wave;
    const int nWaves = gridDim.x * 8;
    const size_t cellOff = (size_t)Bn * STATE;
    const v16bf* bfrag = (const v16bf*)smem_raw;   // 32 v16bf per fragment
    #define BF(g, kc) bfrag[((g) * KC + (kc)) * 32 + lane]

    // Bias is tile-invariant: hoist into registers (per lane: col = jt*16+l15).
    float bIr[8], bOr[8], bCr[8];
    #pragma unroll
    for (int jt = 0; jt < 8; ++jt) {
        bIr[jt] = biasLds[jt * 16 + l15];
        bOr[jt] = biasLds[STATE + jt * 16 + l15];
        bCr[jt] = biasLds[2 * STATE + jt * 16 + l15];
    }

    // Running per-lane base pointers; all intra-pair addressing is immediate.
    const size_t r0 = (size_t)gWave * 32;
    const float* pH  = old_h    + (r0 + l15) * STATE    + hi * 8;  // A rows
    const float* pI  = inp      + (r0 + l15) * IN_FEATS + hi * 8;
    const float* pC  = old_cell + (r0 + hi * 8) * STATE + l15;     // C rows
    float*       pNH = out      + (r0 + hi * 8) * STATE + l15;
    float*       pNC = pNH + cellOff;
    const size_t stepH = (size_t)nWaves * 32 * STATE;
    const size_t stepI = (size_t)nWaves * 32 * IN_FEATS;

    for (int pair = gWave; pair < nPair; pair += nWaves) {
        // ---- Load A fragments for both 16-row halves (documented A layout):
        // lane L: M = L&15; K chunk1 = kc*32 + (L>=16?8:0) .. +7, chunk2 = +16.
        v16bf A0[KC], A1[KC];
        #pragma unroll
        for (int kc = 0; kc < KC; ++kc) {
            const float* s0 = (kc < 4) ? (pH + kc * 32) : pI;
            const float* s1 = (kc < 4) ? (pH + kc * 32 + 16 * STATE)
                                       : (pI + 16 * IN_FEATS);
            union { v16f f; v4f q[4]; } x0, x1;
            x0.q[0] = __builtin_nontemporal_load((const v4f*)(s0));
            x0.q[1] = __builtin_nontemporal_load((const v4f*)(s0 + 4));
            x0.q[2] = __builtin_nontemporal_load((const v4f*)(s0 + 16));
            x0.q[3] = __builtin_nontemporal_load((const v4f*)(s0 + 20));
            x1.q[0] = __builtin_nontemporal_load((const v4f*)(s1));
            x1.q[1] = __builtin_nontemporal_load((const v4f*)(s1 + 4));
            x1.q[2] = __builtin_nontemporal_load((const v4f*)(s1 + 16));
            x1.q[3] = __builtin_nontemporal_load((const v4f*)(s1 + 20));
            A0[kc] = __builtin_convertvector(x0.f, v16bf);
            A1[kc] = __builtin_convertvector(x1.f, v16bf);
        }

        // ---- GEMM: 8 column tiles x 3 gates x 2 row-halves, pipelined B ---
        #pragma unroll
        for (int jt = 0; jt < 8; ++jt) {
            v8f cI0 = {}, cO0 = {}, cC0 = {};   // inline-0 SRC2 on first wmma
            v8f cI1 = {}, cO1 = {}, cC1 = {};

            v16bf bI = BF(jt, 0), bO = BF(8 + jt, 0), bC = BF(16 + jt, 0);
            #pragma unroll
            for (int kc = 0; kc < KC; ++kc) {
                v16bf nI, nO, nC;
                if (kc < KC - 1) {              // prefetch next triple
                    nI = BF(jt,      kc + 1);
                    nO = BF(8 + jt,  kc + 1);
                    nC = BF(16 + jt, kc + 1);
                }
                cI0 = __builtin_amdgcn_wmma_f32_16x16x32_bf16(
                        false, A0[kc], false, bI, (short)0, cI0, false, false);
                cI1 = __builtin_amdgcn_wmma_f32_16x16x32_bf16(
                        false, A1[kc], false, bI, (short)0, cI1, false, false);
                cO0 = __builtin_amdgcn_wmma_f32_16x16x32_bf16(
                        false, A0[kc], false, bO, (short)0, cO0, false, false);
                cO1 = __builtin_amdgcn_wmma_f32_16x16x32_bf16(
                        false, A1[kc], false, bO, (short)0, cO1, false, false);
                cC0 = __builtin_amdgcn_wmma_f32_16x16x32_bf16(
                        false, A0[kc], false, bC, (short)0, cC0, false, false);
                cC1 = __builtin_amdgcn_wmma_f32_16x16x32_bf16(
                        false, A1[kc], false, bC, (short)0, cC1, false, false);
                if (kc < KC - 1) { bI = nI; bO = nO; bC = nC; }
            }

            // C/D layout: VGPR v -> row M = v + (lane>=16 ? 8 : 0), col = lane&15.
            const float bIb = bIr[jt], bOb = bOr[jt], bCb = bCr[jt];
            #pragma unroll
            for (int half = 0; half < 2; ++half) {
                const v8f& cI = half ? cI1 : cI0;
                const v8f& cO = half ? cO1 : cO0;
                const v8f& cC = half ? cC1 : cC0;
                #pragma unroll
                for (int v = 0; v < 8; ++v) {
                    const int off = jt * 16 + v * STATE + half * 16 * STATE;
                    float oc = __builtin_nontemporal_load(pC + off);
                    float ig = fast_sigmoid(cI[v] + bIb);
                    float og = fast_sigmoid(cO[v] + bOb);
                    float cc = cC[v] + bCb;
                    float el = cc > 0.0f ? cc : (__expf(cc) - 1.0f);  // ELU
                    float nc = oc + el * ig;
                    float nh = fast_tanh(nc) * og;
                    __builtin_nontemporal_store(nh, pNH + off);
                    __builtin_nontemporal_store(nc, pNC + off);
                }
            }
        }

        pH += stepH; pI += stepI; pC += stepH; pNH += stepH; pNC += stepH;
    }
    #undef BF
}

extern "C" void kernel_launch(void* const* d_in, const int* in_sizes, int n_in,
                              void* d_out, int out_size, void* d_ws, size_t ws_size,
                              hipStream_t stream) {
    (void)n_in; (void)out_size;
    const float* inp      = (const float*)d_in[0];  // [B,32]
    const float* old_h    = (const float*)d_in[1];  // [B,128]
    const float* old_cell = (const float*)d_in[2];  // [B,128]
    const float* weights  = (const float*)d_in[3];  // [384,160]
    const float* bias     = (const float*)d_in[4];  // [384]
    float* out = (float*)d_out;

    int Bn = in_sizes[1] / STATE;   // batch size

    const bool useWs = ws_size >= (size_t)WS_DWORDS * 4;
    unsigned* pack = useWs ? (unsigned*)d_ws : nullptr;
    if (useWs) {
        dim3 pg((WS_DWORDS + 255) / 256), pb(256);
        lltm_pack_weights<<<pg, pb, 0, stream>>>(weights, bias, pack);
    }

    dim3 grid(256), block(256);
    lltm_fused_wmma<<<grid, block, SMEM_BYTES, stream>>>(
        inp, old_h, old_cell, weights, bias, pack, out, Bn);
}
